// Pangu3DBlock_78469052498458
// MI455X (gfx1250) — matrix operand order repaired and checked by reading, not compiled, not run on hardware
//
#include <hip/hip_runtime.h>
#include <hip/hip_bf16.h>

// ============================================================================
// Pangu/Swin 3D window-attention block, fully fused for gfx1250 (MI455X).
//
//  - wave32, 256 threads/block (8 waves), 2 windows (48 tokens) per block
//  - all GEMMs via v_wmma_f32_16x16x32_f16 (f16 A/B, f32 accum)
//  - weights pre-transposed to N-major f16 in workspace so each B fragment is
//    two 16B loads per lane (k 8h.. and 16+8h..); B frags cached in VGPRs across M-tiles
//  - per-wave loop indices pinned to SGPRs via readfirstlane so tile decode /
//    section branches are scalar (no exec-mask chains)
//  - LayerNorms: one wave per token row, shfl_xor reductions (6 elems/lane)
//  - WMMA fragment layouts per CDNA5 ISA 7.12.2
// ============================================================================

typedef _Float16 v16h __attribute__((ext_vector_type(16)));
typedef _Float16 v8h  __attribute__((ext_vector_type(8)));
typedef float    v8f  __attribute__((ext_vector_type(8)));

#define DIM      192
#define HEADS    6
#define HD       32
#define NTOK     24
#define MLP_HID  768
#define WPB      2            // windows per block
#define MROWS    (WPB*NTOK)   // 48
#define NW_H     24
#define NW_W     16
#define NW_D     7
#define WINS_PER_B (NW_H*NW_W*NW_D)   // 2688
#define NWIN     (2*WINS_PER_B)       // 5376

// workspace byte offsets (16B aligned)
#define WS_QKVWT  0u        // f16 [576][192]  -> 221184 B
#define WS_PROJWT 221184u   // f16 [192][192]  ->  73728 B
#define WS_W1T    294912u   // f16 [768][192]  -> 294912 B
#define WS_W2T    589824u   // f16 [192][768]  -> 294912 B
#define WS_BIAS   884736u   // f32 [6][24][24] ->  13824 B
#define WS_TOTAL  898560u

// ---------------------------------------------------------------------------
// fragment loaders
// ---------------------------------------------------------------------------
__device__ __forceinline__ v16h load_a16(const _Float16* base, int stride) {
  const int l = threadIdx.x & 31;
  const _Float16* p = base + (l & 15) * stride + ((l < 16) ? 0 : 8);
  v8h lo = *(const v8h*)(p);
  v8h hi = *(const v8h*)(p + 16);
  v16h r;
#pragma unroll
  for (int i = 0; i < 8; ++i) { r[i] = lo[i]; r[i + 8] = hi[i]; }
  return r;
}

// B stored N-major: bt[n*stride + k]; fragment = k 8h..8h+7 and 16+8h..16+8h+7 per lane
__device__ __forceinline__ v16h load_b16(const _Float16* bt, int stride) {
  const int l = threadIdx.x & 31;
  const _Float16* p = bt + (l & 15) * stride + ((l < 16) ? 0 : 8);
  v8h lo = *(const v8h*)(p);
  v8h hi = *(const v8h*)(p + 16);
  v16h r;
#pragma unroll
  for (int i = 0; i < 8; ++i) { r[i] = lo[i]; r[i + 8] = hi[i]; }
  return r;
}

__device__ __forceinline__ v8f wmma_f16(v16h a, v16h b, v8f c) {
  return __builtin_amdgcn_wmma_f32_16x16x32_f16(false, a, false, b,
                                                (short)0, c, false, false);
}

__device__ __forceinline__ float wave_sum(float s) {
#pragma unroll
  for (int off = 1; off < 32; off <<= 1) s += __shfl_xor(s, off, 32);
  return s;
}

// ---------------------------------------------------------------------------
// prep: weights f32 -> f16 N-major transpose; gather relative-position bias
// ---------------------------------------------------------------------------
__global__ void pangu_prep_kernel(const float* __restrict__ qkv_w,
                                  const float* __restrict__ proj_w,
                                  const float* __restrict__ mlp_w1,
                                  const float* __restrict__ mlp_w2,
                                  const float* __restrict__ bias_table,
                                  const int*   __restrict__ rel_index,
                                  _Float16* qkv_wt, _Float16* proj_wt,
                                  _Float16* w1t, _Float16* w2t,
                                  float* bias_arr) {
  for (int i = blockIdx.x * blockDim.x + threadIdx.x; i < MLP_HID * DIM;
       i += gridDim.x * blockDim.x) {
    if (i < 3 * DIM * DIM) {            // qkv_wt[n][k] = qkv_w[k][n]
      int n = i / DIM, k = i % DIM;
      qkv_wt[i] = (_Float16)qkv_w[k * (3 * DIM) + n];
    }
    if (i < DIM * DIM) {                // proj_wt[n][k] = proj_w[k][n]
      int n = i / DIM, k = i % DIM;
      proj_wt[i] = (_Float16)proj_w[k * DIM + n];
    }
    if (i < MLP_HID * DIM) {            // w1t[n][k] = mlp_w1[k][n]
      int n = i / DIM, k = i % DIM;
      w1t[i] = (_Float16)mlp_w1[k * MLP_HID + n];
    }
    if (i < DIM * MLP_HID) {            // w2t[n][k] = mlp_w2[k][n]
      int n = i / MLP_HID, k = i % MLP_HID;
      w2t[i] = (_Float16)mlp_w2[k * DIM + n];
    }
    if (i < HEADS * NTOK * NTOK) {      // bias_arr[h][tm][tn]
      int h = i / (NTOK * NTOK), r2 = i % (NTOK * NTOK);
      int tm = r2 / NTOK, tn = r2 % NTOK;
      bias_arr[i] = bias_table[rel_index[tm * NTOK + tn] * HEADS + h];
    }
  }
}

// ---------------------------------------------------------------------------
// fused transformer block: 1 block = 2 windows (48 tokens)
// LDS arena (144KB), phase-overlaid:
//   [0      , 73728 ) : q/k/v f16 [2][6][32][32] each (A/B)  -> s_m [48][768] (D/E)
//   [73728  , 92160 ) : s_h f16 [48][192] (A)         -> s_ao [48][192] (B/C)
//   [92160  ,129024 ) : s_p f16 [8][32][32] (B)       -> s_xw f32 [48][192] (C..E)
//   [129024 ,147456 ) : s_h2 f16 [48][192] (C/D)
// ---------------------------------------------------------------------------
__global__ __launch_bounds__(256)
void pangu_block_kernel(const float* __restrict__ x,
                        const float* __restrict__ ln1_g,
                        const float* __restrict__ ln1_b,
                        const float* __restrict__ qkv_b,
                        const float* __restrict__ proj_b,
                        const float* __restrict__ ln2_g,
                        const float* __restrict__ ln2_b,
                        const float* __restrict__ mlp_b1,
                        const float* __restrict__ mlp_b2,
                        const _Float16* __restrict__ qkv_wt,
                        const _Float16* __restrict__ proj_wt,
                        const _Float16* __restrict__ w1t,
                        const _Float16* __restrict__ w2t,
                        const float* __restrict__ bias_arr,
                        float* __restrict__ out) {
  __shared__ __align__(16) unsigned char smem[147456];
  __shared__ int s_rowoff[MROWS];

  _Float16* s_q  = (_Float16*)(smem);            // [2*6][32][32]
  _Float16* s_k  = (_Float16*)(smem) + 12288;
  _Float16* s_v  = (_Float16*)(smem) + 24576;    // stored [dim][tok]
  _Float16* s_m  = (_Float16*)(smem);            // [48][768]
  _Float16* s_h  = (_Float16*)(smem + 73728);    // [48][192]
  _Float16* s_ao = (_Float16*)(smem + 73728);    // [48][192]
  _Float16* s_p  = (_Float16*)(smem + 92160);    // [8][32][32]
  float*    s_xw = (float*)   (smem + 92160);    // [48][192]
  _Float16* s_h2 = (_Float16*)(smem + 129024);   // [48][192]

  const int tid  = threadIdx.x;
  const int wave = __builtin_amdgcn_readfirstlane(tid >> 5);
  const int lane = tid & 31;
  const int col  = lane & 15;
  const int rb   = (lane < 16) ? 0 : 8;

  // zero q/k/v region (pad rows/cols 24..31 must be 0)
  {
    uint32_t* p = (uint32_t*)smem;
    for (int i = tid; i < 73728 / 4; i += 256) p[i] = 0u;
  }
  // token -> global element offset (window gather)
  if (tid < MROWS) {
    int wl = tid / NTOK, t = tid % NTOK;
    int win = blockIdx.x * WPB + wl;
    int b = win / WINS_PER_B, r = win % WINS_PER_B;
    int hq = r / (NW_W * NW_D), wq = (r / NW_D) % NW_W, dq = r % NW_D;
    int i = t / 12, j = (t % 12) / 2, kk = t & 1;
    int gh = hq * 2 + i, gw = wq * 6 + j, gd = dq * 2 + kk;
    s_rowoff[tid] = (((b * 48 + gh) * 96 + gw) * 14 + gd) * DIM;
  }
  __syncthreads();

  // ---- Phase A: LN1 -> s_h (f16); one wave per row -------------------------
  for (int row = wave; row < MROWS; row += 8) {
    const float* xp = x + s_rowoff[row];
    float vbuf[6];
    float s = 0.f;
#pragma unroll
    for (int q = 0; q < 6; ++q) { vbuf[q] = xp[lane + 32 * q]; s += vbuf[q]; }
    float mu = wave_sum(s) * (1.f / DIM);
    float vs = 0.f;
#pragma unroll
    for (int q = 0; q < 6; ++q) { float d = vbuf[q] - mu; vs += d * d; }
    float rs = rsqrtf(wave_sum(vs) * (1.f / DIM) + 1e-5f);
#pragma unroll
    for (int q = 0; q < 6; ++q) {
      int c = lane + 32 * q;
      s_h[row * DIM + c] = (_Float16)((vbuf[q] - mu) * rs * ln1_g[c] + ln1_b[c]);
    }
  }
  __syncthreads();

  // ---- Phase A2: QKV GEMM [48x192]@[192x576]; wave owns an N-strip ---------
  for (int nt0 = wave; nt0 < 36; nt0 += 8) {
    const int nt = __builtin_amdgcn_readfirstlane(nt0);   // wave-uniform -> SGPR
    v16h bfr[6];
#pragma unroll
    for (int ks = 0; ks < 6; ++ks)
      bfr[ks] = load_b16(qkv_wt + (nt * 16) * DIM + ks * 32, DIM);
    // tile-uniform decode: 16-wide N tile never straddles section/head
    const int sec  = nt / 12;                // 0=q,1=k,2=v  (scalar)
    const int head = (nt % 12) >> 1;         // 0..5         (scalar)
    const int dim  = ((nt & 1) << 4) + col;  // 0..31
    const float bias = qkv_b[nt * 16 + col];
    for (int mt = 0; mt < 3; ++mt) {
      v8f acc = {};
#pragma unroll
      for (int ks = 0; ks < 6; ++ks)
        acc = wmma_f16(load_a16(s_h + mt * 16 * DIM + ks * 32, DIM), bfr[ks], acc);
      if (sec == 0) {
#pragma unroll
        for (int r = 0; r < 8; ++r) {
          int row = mt * 16 + rb + r;
          int wl = (row >= NTOK) ? 1 : 0, tok = row - wl * NTOK;
          s_q[(wl * HEADS + head) * 1024 + tok * 32 + dim] =
              (_Float16)((acc[r] + bias) * 0.17677669529663687f);
        }
      } else if (sec == 1) {
#pragma unroll
        for (int r = 0; r < 8; ++r) {
          int row = mt * 16 + rb + r;
          int wl = (row >= NTOK) ? 1 : 0, tok = row - wl * NTOK;
          s_k[(wl * HEADS + head) * 1024 + tok * 32 + dim] = (_Float16)(acc[r] + bias);
        }
      } else {
#pragma unroll
        for (int r = 0; r < 8; ++r) {
          int row = mt * 16 + rb + r;
          int wl = (row >= NTOK) ? 1 : 0, tok = row - wl * NTOK;
          s_v[(wl * HEADS + head) * 1024 + dim * 32 + tok] = (_Float16)(acc[r] + bias);
        }
      }
    }
  }
  __syncthreads();

  // ---- Phase B: attention per (window, head) -------------------------------
  _Float16* myp = s_p + wave * 1024;
  for (int u0 = wave; u0 < WPB * HEADS; u0 += 8) {
    const int u = __builtin_amdgcn_readfirstlane(u0);
    const int wl = (u >= HEADS) ? 1 : 0, h = u - wl * HEADS;  // scalar
    const _Float16* qb = s_q + u * 1024;
    const _Float16* kb = s_k + u * 1024;   // [tok][dim] == (K^T) N-major
    const _Float16* vb = s_v + u * 1024;   // [dim][tok] == V N-major
    const float* bt = bias_arr + h * NTOK * NTOK;

    v8f S[2][2];
    for (int mt = 0; mt < 2; ++mt) {
      v16h a = load_a16(qb + mt * 16 * 32, 32);
      for (int nt = 0; nt < 2; ++nt) {
        v16h b = load_b16(kb + nt * 16 * 32, 32);
        v8f z = {};
        S[mt][nt] = wmma_f16(a, b, z);
      }
    }
    // softmax rows (cols 0..23 valid), stage P (f16) per-wave
    for (int mt = 0; mt < 2; ++mt) {
#pragma unroll
      for (int r = 0; r < 8; ++r) {
        int tokm = mt * 16 + rb + r;
        float v0 = S[mt][0][r], v1 = S[mt][1][r];
        int tn1 = 16 + col;
        if (tokm < NTOK) {
          v0 += bt[tokm * NTOK + col];
          if (tn1 < NTOK) v1 += bt[tokm * NTOK + tn1];
        }
        if (tn1 >= NTOK) v1 = -1e30f;
        float m = fmaxf(v0, v1);
#pragma unroll
        for (int off = 1; off < 16; off <<= 1) m = fmaxf(m, __shfl_xor(m, off, 32));
        float e0 = __expf(v0 - m);
        float e1 = (tn1 < NTOK) ? __expf(v1 - m) : 0.f;
        float s = e0 + e1;
#pragma unroll
        for (int off = 1; off < 16; off <<= 1) s += __shfl_xor(s, off, 32);
        float inv = 1.f / s;
        myp[tokm * 32 + col]      = (_Float16)(e0 * inv);
        myp[tokm * 32 + 16 + col] = (_Float16)(e1 * inv);
      }
    }
    // O = P @ V  -> s_ao[tok][head*32+dim]
    for (int mt = 0; mt < 2; ++mt) {
      v16h a = load_a16(myp + mt * 16 * 32, 32);
      for (int nt = 0; nt < 2; ++nt) {
        v16h b = load_b16(vb + nt * 16 * 32, 32);
        v8f z = {};
        v8f o = wmma_f16(a, b, z);
#pragma unroll
        for (int r = 0; r < 8; ++r) {
          int tokm = mt * 16 + rb + r;
          if (tokm < NTOK)
            s_ao[(wl * NTOK + tokm) * DIM + h * HD + nt * 16 + col] = (_Float16)o[r];
        }
      }
    }
  }
  __syncthreads();

  // ---- Phase C: proj GEMM + residual + LN2 ---------------------------------
  for (int nt0 = wave; nt0 < 12; nt0 += 8) {
    const int nt = __builtin_amdgcn_readfirstlane(nt0);
    v16h bfr[6];
#pragma unroll
    for (int ks = 0; ks < 6; ++ks)
      bfr[ks] = load_b16(proj_wt + (nt * 16) * DIM + ks * 32, DIM);
    const float pb = proj_b[nt * 16 + col];
    for (int mt = 0; mt < 3; ++mt) {
      v8f acc = {};
#pragma unroll
      for (int ks = 0; ks < 6; ++ks)
        acc = wmma_f16(load_a16(s_ao + mt * 16 * DIM + ks * 32, DIM), bfr[ks], acc);
#pragma unroll
      for (int r = 0; r < 8; ++r)
        s_xw[(mt * 16 + rb + r) * DIM + nt * 16 + col] = acc[r] + pb;
    }
  }
  __syncthreads();
  // residual + LN2; one wave per row
  for (int row = wave; row < MROWS; row += 8) {
    const float* xp = x + s_rowoff[row];
    float* xwr = s_xw + row * DIM;
    float vbuf[6];
    float s = 0.f;
#pragma unroll
    for (int q = 0; q < 6; ++q) {
      int c = lane + 32 * q;
      float v = xp[c] + xwr[c];
      vbuf[q] = v; s += v;
    }
    float mu = wave_sum(s) * (1.f / DIM);
    float vs = 0.f;
#pragma unroll
    for (int q = 0; q < 6; ++q) { float d = vbuf[q] - mu; vs += d * d; }
    float rs = rsqrtf(wave_sum(vs) * (1.f / DIM) + 1e-5f);
#pragma unroll
    for (int q = 0; q < 6; ++q) {
      int c = lane + 32 * q;
      xwr[c] = vbuf[q];
      s_h2[row * DIM + c] = (_Float16)((vbuf[q] - mu) * rs * ln2_g[c] + ln2_b[c]);
    }
  }
  __syncthreads();

  // ---- Phase D: MLP1 [48x192]@[192x768] + GELU -> s_m (f16) ----------------
  for (int nt0 = wave; nt0 < 48; nt0 += 8) {
    const int nt = __builtin_amdgcn_readfirstlane(nt0);
    v16h bfr[6];
#pragma unroll
    for (int ks = 0; ks < 6; ++ks)
      bfr[ks] = load_b16(w1t + (nt * 16) * DIM + ks * 32, DIM);
    const float b1 = mlp_b1[nt * 16 + col];
    for (int mt = 0; mt < 3; ++mt) {
      v8f acc = {};
#pragma unroll
      for (int ks = 0; ks < 6; ++ks)
        acc = wmma_f16(load_a16(s_h2 + mt * 16 * DIM + ks * 32, DIM), bfr[ks], acc);
#pragma unroll
      for (int r = 0; r < 8; ++r) {
        float vv = acc[r] + b1;
        vv = 0.5f * vv * (1.f + erff(vv * 0.7071067811865476f));
        s_m[(mt * 16 + rb + r) * MLP_HID + nt * 16 + col] = (_Float16)vv;
      }
    }
  }
  __syncthreads();

  // ---- Phase E: MLP2 [48x768]@[768x192] + residual -> out (scatter) --------
  for (int t0 = wave; t0 < 3 * 12; t0 += 8) {
    const int t = __builtin_amdgcn_readfirstlane(t0);
    const int mt = (t >= 24) ? 2 : ((t >= 12) ? 1 : 0);   // scalar
    const int nt = t - mt * 12;                           // scalar
    v8f acc = {};
    for (int ks = 0; ks < 24; ++ks) {
      v16h a = load_a16(s_m + mt * 16 * MLP_HID + ks * 32, MLP_HID);
      v16h b = load_b16(w2t + (nt * 16) * MLP_HID + ks * 32, MLP_HID);
      acc = wmma_f16(a, b, acc);
    }
    const int n = nt * 16 + col;
    const float b2 = mlp_b2[n];
#pragma unroll
    for (int r = 0; r < 8; ++r) {
      int row = mt * 16 + rb + r;
      out[s_rowoff[row] + n] = s_xw[row * DIM + n] + acc[r] + b2;
    }
  }
}

// ---------------------------------------------------------------------------
extern "C" void kernel_launch(void* const* d_in, const int* in_sizes, int n_in,
                              void* d_out, int out_size, void* d_ws, size_t ws_size,
                              hipStream_t stream) {
  const float* x          = (const float*)d_in[0];
  const float* ln1_g      = (const float*)d_in[1];
  const float* ln1_b      = (const float*)d_in[2];
  const float* qkv_w      = (const float*)d_in[3];
  const float* qkv_b      = (const float*)d_in[4];
  const float* bias_table = (const float*)d_in[5];
  const float* proj_w     = (const float*)d_in[6];
  const float* proj_b     = (const float*)d_in[7];
  const float* ln2_g      = (const float*)d_in[8];
  const float* ln2_b      = (const float*)d_in[9];
  const float* mlp_w1     = (const float*)d_in[10];
  const float* mlp_b1     = (const float*)d_in[11];
  const float* mlp_w2     = (const float*)d_in[12];
  const float* mlp_b2     = (const float*)d_in[13];
  const int*   rel_index  = (const int*)d_in[14];

  unsigned char* ws = (unsigned char*)d_ws;
  _Float16* qkv_wt  = (_Float16*)(ws + WS_QKVWT);
  _Float16* proj_wt = (_Float16*)(ws + WS_PROJWT);
  _Float16* w1t     = (_Float16*)(ws + WS_W1T);
  _Float16* w2t     = (_Float16*)(ws + WS_W2T);
  float*    bias_arr= (float*)   (ws + WS_BIAS);

  pangu_prep_kernel<<<(MLP_HID * DIM + 255) / 256, 256, 0, stream>>>(
      qkv_w, proj_w, mlp_w1, mlp_w2, bias_table, rel_index,
      qkv_wt, proj_wt, w1t, w2t, bias_arr);

  pangu_block_kernel<<<NWIN / WPB, 256, 0, stream>>>(
      x, ln1_g, ln1_b, qkv_b, proj_b, ln2_g, ln2_b, mlp_b1, mlp_b2,
      qkv_wt, proj_wt, w1t, w2t, bias_arr, (float*)d_out);
}
